// MHAttention_83511344103698
// MI455X (gfx1250) — matrix operand order repaired and checked by reading, not compiled, not run on hardware
//
#include <hip/hip_runtime.h>
#include <hip/hip_bf16.h>

#define S_LEN   2048
#define DMODEL  2048
#define NHEADS  32
#define NKVH    8
#define HDIM    64
#define NTOK    4096   // B * S

typedef __attribute__((ext_vector_type(16))) __bf16 v16bf;
typedef __attribute__((ext_vector_type(8)))  __bf16 v8bf;
typedef __attribute__((ext_vector_type(8)))  float  v8f;
typedef __attribute__((ext_vector_type(4)))  unsigned int u32x4;
typedef __attribute__((ext_vector_type(8)))  unsigned int u32x8;

__device__ __forceinline__ unsigned short f2bf(float f) {
  unsigned int u = __builtin_bit_cast(unsigned int, f);
  u += 0x7FFFu + ((u >> 16) & 1u);           // round-to-nearest-even
  return (unsigned short)(u >> 16);
}

__device__ __forceinline__ v16bf ldfrag(const unsigned short* p0,
                                        const unsigned short* p1) {
  v8bf lo = *(const v8bf*)p0;
  v8bf hi = *(const v8bf*)p1;
  return __builtin_shufflevector(lo, hi, 0,1,2,3,4,5,6,7,8,9,10,11,12,13,14,15);
}

// ---------------------------------------------------------------- TDM tile load
// DMA a 128x32 bf16 tile (row stride = k elements in memory) into LDS with
// 4-DWORD padding after each 16-DWORD (64B) row -> LDS row stride 40 elements.
__device__ __forceinline__ void tdm_load_tile(unsigned lds_addr,
                                              unsigned long long gaddr,
                                              unsigned k_elems) {
  u32x4 g0;
  g0[0] = 1u;                                          // count=1, no gather
  g0[1] = lds_addr;                                    // lds_addr [63:32]
  g0[2] = (unsigned)gaddr;                             // global_addr lo
  g0[3] = ((unsigned)(gaddr >> 32) & 0x01FFFFFFu)      // global_addr [56:32]
          | (2u << 30);                                // type = 2 (image)
  u32x8 g1;
  g1[0] = (1u << 16)      // data_size = 2 bytes
        | (1u << 20)      // pad_enable
        | (3u << 22)      // pad_interval: 16 DWORDs (one 64B tile row)
        | (3u << 25);     // pad_amount: 4 DWORDs (16B -> stride 40 elems)
  g1[1] = (1u << 20) << 16;            // tensor_dim0[15:0] in [63:48] (large)
  g1[2] = ((1u << 20) >> 16)           // tensor_dim0[31:16]
        | (0u << 16);                  // tensor_dim1[15:0]
  g1[3] = (1u << 4)                    // tensor_dim1[31:16] (large)
        | (32u << 16);                 // tile_dim0 = 32 elems (K)
  g1[4] = 128u;                        // tile_dim1 = 128 rows, tile_dim2 = 0
  g1[5] = k_elems;                     // tensor_dim0_stride lo (elements)
  g1[6] = 0u;                          // stride0 hi, dim1_stride lo
  g1[7] = 0u;
  asm volatile("tensor_load_to_lds %0, %1" : : "s"(g0), "s"(g1) : "memory");
}

// ---------------------------------------------------------------- convert
__global__ void cvt_f32_to_bf16(const float* __restrict__ src,
                                unsigned short* __restrict__ dst, int n) {
  int i = blockIdx.x * blockDim.x + threadIdx.x;
  if (i < n) dst[i] = f2bf(src[i]);
}

// ---------------------------------------------------------------- GEMM
// C[M,N] = A[M,K] @ B[N,K]^T   (A,B bf16 row-major, C f32)
// TDM double-buffered staging: wave 0 issues tensor_load_to_lds for step i+1
// while all waves run WMMAs on step i; s_wait_tensorcnt 2 retires the current
// buffer's DMA pair while the next pair stays in flight.
#define BM 128
#define BN 128
#define BK 32

__global__ __launch_bounds__(256)
void gemm_bf16_wmma(const unsigned short* __restrict__ A,
                    const unsigned short* __restrict__ Bw,
                    float* __restrict__ C, int M, int N, int K) {
  __shared__ unsigned short sA[2][BM][BK + 8];   // stride 40 elems (80B)
  __shared__ unsigned short sB[2][BN][BK + 8];

  const int tid  = threadIdx.x;
  const int wave = tid >> 5;
  const int lane = tid & 31;
  const int r    = lane & 15;
  const int hl   = lane >> 4;
  const int wm   = wave >> 2;   // 0..1
  const int wn   = wave & 3;    // 0..3

  const int rowBase = blockIdx.y * BM;
  const int colBase = blockIdx.x * BN;

  const unsigned long long aBase =
      (unsigned long long)(const void*)A + (size_t)rowBase * K * 2ull;
  const unsigned long long bBase =
      (unsigned long long)(const void*)Bw + (size_t)colBase * K * 2ull;

  v8f zero = {};
  v8f acc[4][2];
#pragma unroll
  for (int i = 0; i < 4; ++i)
#pragma unroll
    for (int j = 0; j < 2; ++j) acc[i][j] = zero;

  const int nk = K / BK;
  if (wave == 0) {   // prologue: DMA k-step 0 into buffer 0
    tdm_load_tile((unsigned)(size_t)&sA[0][0][0], aBase, (unsigned)K);
    tdm_load_tile((unsigned)(size_t)&sB[0][0][0], bBase, (unsigned)K);
  }

  for (int i = 0; i < nk; ++i) {
    __syncthreads();               // readers of buf[(i+1)&1] (iter i-1) done
    if (wave == 0) {
      if (i + 1 < nk) {
        const unsigned long long off = (unsigned long long)(i + 1) * BK * 2ull;
        tdm_load_tile((unsigned)(size_t)&sA[(i + 1) & 1][0][0], aBase + off, (unsigned)K);
        tdm_load_tile((unsigned)(size_t)&sB[(i + 1) & 1][0][0], bBase + off, (unsigned)K);
        __builtin_amdgcn_s_wait_tensorcnt(2);   // current buffer's pair done
      } else {
        __builtin_amdgcn_s_wait_tensorcnt(0);
      }
    }
    __syncthreads();               // buffer i&1 visible to all waves

    const int buf = i & 1;
    v16bf afr[4], bfr[2];
#pragma unroll
    for (int mi = 0; mi < 4; ++mi) {
      const int row = wm * 64 + mi * 16 + r;
      // A 16x32 bf16 layout: per lane two contiguous 8-elem K chunks
      afr[mi] = ldfrag(&sA[buf][row][hl * 8], &sA[buf][row][hl * 8 + 16]);
    }
#pragma unroll
    for (int ni = 0; ni < 2; ++ni) {
      const int col = wn * 32 + ni * 16 + r;
      // B 32x16 bf16 layout: per lane two contiguous 8-elem K chunks (same as A)
      bfr[ni] = ldfrag(&sB[buf][col][hl * 8], &sB[buf][col][hl * 8 + 16]);
    }
#pragma unroll
    for (int mi = 0; mi < 4; ++mi)
#pragma unroll
      for (int ni = 0; ni < 2; ++ni)
        acc[mi][ni] = __builtin_amdgcn_wmma_f32_16x16x32_bf16(
            false, afr[mi], false, bfr[ni], (short)0, acc[mi][ni], false, false);
  }

#pragma unroll
  for (int mi = 0; mi < 4; ++mi)
#pragma unroll
    for (int ni = 0; ni < 2; ++ni) {
      const int col = colBase + wn * 32 + ni * 16 + r;
#pragma unroll
      for (int e = 0; e < 8; ++e) {
        const int row = rowBase + wm * 64 + mi * 16 + hl * 8 + e;
        C[(size_t)row * N + col] = acc[mi][ni][e];
      }
    }
}

// ---------------------------------------------------------------- RMSNorm + RoPE
// qf [NTOK,2048], kf/vf [NTOK,512]  ->  head-major bf16 [B,H,S,64]
__global__ __launch_bounds__(256)
void normrope_kernel(const float* __restrict__ qf, const float* __restrict__ kf,
                     const float* __restrict__ vf,
                     const float* __restrict__ qw, const float* __restrict__ kw,
                     unsigned short* __restrict__ qbf, unsigned short* __restrict__ kbf,
                     unsigned short* __restrict__ vbf) {
  const int wave = threadIdx.x >> 5;
  const int lane = threadIdx.x & 31;
  const int task = blockIdx.x * 8 + wave;
  const int token = task / 48;
  const int unit  = task % 48;      // 0..31 q, 32..39 k, 40..47 v
  const int b     = token >> 11;    // / 2048
  const int spos  = token & 2047;

  if (unit < 40) {
    const bool isQ = unit < 32;
    const int head = isQ ? unit : unit - 32;
    const float* src = isQ ? (qf + (size_t)token * DMODEL + head * HDIM)
                           : (kf + (size_t)token * (NKVH * HDIM) + head * HDIM);
    float x1 = src[lane];
    float x2 = src[lane + 32];
    float ss = x1 * x1 + x2 * x2;
    ss += __shfl_xor(ss, 1);
    ss += __shfl_xor(ss, 2);
    ss += __shfl_xor(ss, 4);
    ss += __shfl_xor(ss, 8);
    ss += __shfl_xor(ss, 16);
    const float rs = rsqrtf(ss * (1.0f / 64.0f) + 1e-6f);
    const float* w = isQ ? qw : kw;
    const float xn1 = x1 * rs * w[lane];
    const float xn2 = x2 * rs * w[lane + 32];
    // rope: inv_freq = 10000^(-lane/32)
    const float ang = (float)spos * __expf(-(float)lane * 0.28782313662425574f);
    float sn, cs;
    __sincosf(ang, &sn, &cs);
    const float o1 = xn1 * cs - xn2 * sn;
    const float o2 = xn2 * cs + xn1 * sn;
    unsigned short* dst =
        isQ ? (qbf + ((size_t)(b * NHEADS + head) * S_LEN + spos) * HDIM)
            : (kbf + ((size_t)(b * NKVH + head) * S_LEN + spos) * HDIM);
    dst[lane]      = f2bf(o1);
    dst[lane + 32] = f2bf(o2);
  } else {
    const int head = unit - 40;
    const float* src = vf + (size_t)token * (NKVH * HDIM) + head * HDIM;
    unsigned short* dst = vbf + ((size_t)(b * NKVH + head) * S_LEN + spos) * HDIM;
    dst[lane]      = f2bf(src[lane]);
    dst[lane + 32] = f2bf(src[lane + 32]);
  }
}

// ---------------------------------------------------------------- flash attention
// Qb [B*H, S, 64], Kb/Vb [B*KVH, S, 64] bf16 -> Ob [B*S, H*64] bf16
__global__ __launch_bounds__(128)
void attn_fwd_wmma(const unsigned short* __restrict__ Qb,
                   const unsigned short* __restrict__ Kb,
                   const unsigned short* __restrict__ Vb,
                   unsigned short* __restrict__ Ob) {
  __shared__ unsigned short sK [32][72];   // key-major, 144B stride (16B aligned)
  __shared__ unsigned short sVt[64][40];   // dim-major (transposed V), 80B stride
  __shared__ unsigned short sP [4][16][40];

  const int tid  = threadIdx.x;
  const int wave = tid >> 5;
  const int lane = tid & 31;
  const int r    = lane & 15;
  const int hl   = lane >> 4;

  const int bh  = blockIdx.y;          // b*NHEADS + h
  const int b   = bh >> 5;
  const int h   = bh & 31;
  const int kvh = h >> 2;              // 4 q heads per kv head
  const int qBlock = blockIdx.x;
  const int qRow0  = qBlock * 64 + wave * 16;

  // resident Q fragments (16 rows x 64 dims = two 16x32 A tiles)
  v16bf qfr[2];
  {
    const unsigned short* qp = Qb + ((size_t)bh * S_LEN + qRow0 + r) * HDIM;
#pragma unroll
    for (int ks = 0; ks < 2; ++ks)
      qfr[ks] = ldfrag(qp + ks * 32 + hl * 8, qp + ks * 32 + hl * 8 + 16);
  }

  v8f zero = {};
  float m[8], l[8];
  v8f o[4];
#pragma unroll
  for (int e = 0; e < 8; ++e) { m[e] = -1e30f; l[e] = 0.0f; }
#pragma unroll
  for (int ni = 0; ni < 4; ++ni) o[ni] = zero;

  const unsigned short* kBase = Kb + (size_t)(b * NKVH + kvh) * S_LEN * HDIM;
  const unsigned short* vBase = Vb + (size_t)(b * NKVH + kvh) * S_LEN * HDIM;
  const int nKb = qBlock * 2 + 2;      // key blocks of 32 keys, causal bound

  for (int kblk = 0; kblk < nKb; ++kblk) {
    const int key0 = kblk * 32;
    {   // cooperative stage: K key-major, V dim-major (transposed)
      const int row = tid >> 2;            // 0..31 (key)
      const int q4  = (tid & 3) * 16;      // dim chunk
      const unsigned short* gk = kBase + (size_t)(key0 + row) * HDIM + q4;
      const unsigned short* gv = vBase + (size_t)(key0 + row) * HDIM + q4;
      uint4 kk0 = *(const uint4*)gk;
      uint4 kk1 = *(const uint4*)(gk + 8);
      union { uint4 u; unsigned short s[8]; } vv0, vv1;
      vv0.u = *(const uint4*)gv;
      vv1.u = *(const uint4*)(gv + 8);
      __syncthreads();
      *(uint4*)&sK[row][q4]     = kk0;
      *(uint4*)&sK[row][q4 + 8] = kk1;
#pragma unroll
      for (int j = 0; j < 8; ++j) {
        sVt[q4 + j][row]     = vv0.s[j];
        sVt[q4 + 8 + j][row] = vv1.s[j];
      }
      __syncthreads();
    }

    // scores: S(16x32) = Q(16x64) @ K^T(64x32)
    v8f sacc[2];
    sacc[0] = zero; sacc[1] = zero;
#pragma unroll
    for (int ni = 0; ni < 2; ++ni) {
      const int keyc = ni * 16 + r;
#pragma unroll
      for (int ks = 0; ks < 2; ++ks) {
        v16bf bk = ldfrag(&sK[keyc][ks * 32 + hl * 8],
                          &sK[keyc][ks * 32 + hl * 8 + 16]);
        sacc[ni] = __builtin_amdgcn_wmma_f32_16x16x32_bf16(
            false, qfr[ks], false, bk, (short)0, sacc[ni], false, false);
      }
    }

    // scale + causal mask
    const int qrowBase = qRow0 + hl * 8;
#pragma unroll
    for (int ni = 0; ni < 2; ++ni) {
      const int scol = key0 + ni * 16 + r;
#pragma unroll
      for (int e = 0; e < 8; ++e) {
        float v = sacc[ni][e] * 0.125f;        // HEAD_DIM^-0.5
        if (scol > qrowBase + e) v = -1e30f;
        sacc[ni][e] = v;
      }
    }

    // online softmax (row reductions inside each 16-lane half)
#pragma unroll
    for (int e = 0; e < 8; ++e) {
      float v = fmaxf(sacc[0][e], sacc[1][e]);
      v = fmaxf(v, __shfl_xor(v, 1));
      v = fmaxf(v, __shfl_xor(v, 2));
      v = fmaxf(v, __shfl_xor(v, 4));
      v = fmaxf(v, __shfl_xor(v, 8));
      const float mn = fmaxf(m[e], v);
      const float alpha = __expf(m[e] - mn);
      m[e] = mn;
      l[e] *= alpha;
#pragma unroll
      for (int ni = 0; ni < 4; ++ni) o[ni][e] *= alpha;
      float rs = 0.0f;
#pragma unroll
      for (int ni = 0; ni < 2; ++ni) {
        const float p = __expf(sacc[ni][e] - mn);
        sacc[ni][e] = p;
        rs += p;
      }
      rs += __shfl_xor(rs, 1);
      rs += __shfl_xor(rs, 2);
      rs += __shfl_xor(rs, 4);
      rs += __shfl_xor(rs, 8);
      l[e] += rs;
    }

    // C-layout -> A-layout via per-wave LDS round trip (same-wave DS is in-order)
#pragma unroll
    for (int ni = 0; ni < 2; ++ni) {
      const int col = ni * 16 + r;
#pragma unroll
      for (int e = 0; e < 8; ++e)
        sP[wave][hl * 8 + e][col] = f2bf(sacc[ni][e]);
    }
    const unsigned short* pp = &sP[wave][r][0];
    v16bf pf = ldfrag(pp + hl * 8, pp + hl * 8 + 16);

    // O(16x64) += P(16x32) @ V(32x64)
#pragma unroll
    for (int ni = 0; ni < 4; ++ni) {
      const int dcol = ni * 16 + r;
      v16bf bv = ldfrag(&sVt[dcol][hl * 8], &sVt[dcol][hl * 8 + 16]);
      o[ni] = __builtin_amdgcn_wmma_f32_16x16x32_bf16(
          false, pf, false, bv, (short)0, o[ni], false, false);
    }
  }

  // normalize + store bf16 into [B*S, H*64]
#pragma unroll
  for (int e = 0; e < 8; ++e) l[e] = 1.0f / l[e];
#pragma unroll
  for (int ni = 0; ni < 4; ++ni) {
    const int col = h * HDIM + ni * 16 + r;
#pragma unroll
    for (int e = 0; e < 8; ++e) {
      const int qrow = qRow0 + hl * 8 + e;
      Ob[(size_t)(b * S_LEN + qrow) * DMODEL + col] = f2bf(o[ni][e] * l[e]);
    }
  }
}

// ---------------------------------------------------------------- launcher
extern "C" void kernel_launch(void* const* d_in, const int* in_sizes, int n_in,
                              void* d_out, int out_size, void* d_ws, size_t ws_size,
                              hipStream_t stream) {
  (void)in_sizes; (void)n_in; (void)out_size; (void)ws_size;
  const float* x   = (const float*)d_in[0];
  const float* wq  = (const float*)d_in[1];
  const float* wk  = (const float*)d_in[2];
  const float* wv  = (const float*)d_in[3];
  const float* wo  = (const float*)d_in[4];
  const float* qnw = (const float*)d_in[5];
  const float* knw = (const float*)d_in[6];
  float* out = (float*)d_out;

  char* ws = (char*)d_ws;
  unsigned short* xb   = (unsigned short*)ws; ws += (size_t)NTOK * DMODEL * 2;
  unsigned short* wqb  = (unsigned short*)ws; ws += (size_t)DMODEL * DMODEL * 2;
  unsigned short* wkb  = (unsigned short*)ws; ws += (size_t)(NKVH*HDIM) * DMODEL * 2;
  unsigned short* wvb  = (unsigned short*)ws; ws += (size_t)(NKVH*HDIM) * DMODEL * 2;
  unsigned short* wob  = (unsigned short*)ws; ws += (size_t)DMODEL * DMODEL * 2;
  float* qf            = (float*)ws;          ws += (size_t)NTOK * DMODEL * 4;
  float* kf            = (float*)ws;          ws += (size_t)NTOK * (NKVH*HDIM) * 4;
  float* vf            = (float*)ws;          ws += (size_t)NTOK * (NKVH*HDIM) * 4;
  unsigned short* qbf  = (unsigned short*)ws; ws += (size_t)NTOK * DMODEL * 2;
  unsigned short* kbf  = (unsigned short*)ws; ws += (size_t)NTOK * (NKVH*HDIM) * 2;
  unsigned short* vbf  = (unsigned short*)ws; ws += (size_t)NTOK * (NKVH*HDIM) * 2;
  unsigned short* attn = (unsigned short*)ws; ws += (size_t)NTOK * DMODEL * 2;

  const int T = 256;
  // converts
  cvt_f32_to_bf16<<<(NTOK*DMODEL + T-1)/T, T, 0, stream>>>(x,  xb,  NTOK*DMODEL);
  cvt_f32_to_bf16<<<(DMODEL*DMODEL + T-1)/T, T, 0, stream>>>(wq, wqb, DMODEL*DMODEL);
  cvt_f32_to_bf16<<<(NKVH*HDIM*DMODEL + T-1)/T, T, 0, stream>>>(wk, wkb, NKVH*HDIM*DMODEL);
  cvt_f32_to_bf16<<<(NKVH*HDIM*DMODEL + T-1)/T, T, 0, stream>>>(wv, wvb, NKVH*HDIM*DMODEL);
  cvt_f32_to_bf16<<<(DMODEL*DMODEL + T-1)/T, T, 0, stream>>>(wo, wob, DMODEL*DMODEL);

  // QKV projections
  gemm_bf16_wmma<<<dim3(DMODEL/BN, NTOK/BM), 256, 0, stream>>>(xb, wqb, qf, NTOK, DMODEL, DMODEL);
  gemm_bf16_wmma<<<dim3((NKVH*HDIM)/BN, NTOK/BM), 256, 0, stream>>>(xb, wkb, kf, NTOK, NKVH*HDIM, DMODEL);
  gemm_bf16_wmma<<<dim3((NKVH*HDIM)/BN, NTOK/BM), 256, 0, stream>>>(xb, wvb, vf, NTOK, NKVH*HDIM, DMODEL);

  // rmsnorm + rope + bf16 repack (48 wave-tasks per token)
  normrope_kernel<<<(NTOK*48)/8, 256, 0, stream>>>(qf, kf, vf, qnw, knw, qbf, kbf, vbf);

  // flash attention
  attn_fwd_wmma<<<dim3(S_LEN/64, 2*NHEADS), 128, 0, stream>>>(qbf, kbf, vbf, attn);

  // output projection (f32 out)
  gemm_bf16_wmma<<<dim3(DMODEL/BN, NTOK/BM), 256, 0, stream>>>(attn, wob, out, NTOK, DMODEL, DMODEL);
}
